// RNNTJoint_40638980555333
// MI455X (gfx1250) — compile-verified, matrix operand order checked by reading, not run on hardware
//
#include <hip/hip_runtime.h>

typedef _Float16 v16h __attribute__((ext_vector_type(16)));
typedef _Float16 v8h  __attribute__((ext_vector_type(8)));
typedef float    v8f  __attribute__((ext_vector_type(8)));

#define WMMA_F16(a, b, c) \
  __builtin_amdgcn_wmma_f32_16x16x32_f16(false, (a), false, (b), (short)0, (c), false, false)

// Fragment from two 16B chunks at p and p+16 halves (row-major source).
__device__ __forceinline__ v16h make_frag(const _Float16* p) {
  v8h lo = *(const v8h*)(p);
  v8h hi = *(const v8h*)(p + 16);
  return __builtin_shufflevector(lo, hi, 0,1,2,3,4,5,6,7,8,9,10,11,12,13,14,15);
}

// Fragment from a packed 32B per-lane blob (lo at p, hi at p+8 halves).
__device__ __forceinline__ v16h make_frag_packed(const _Float16* p) {
  v8h lo = *(const v8h*)(p);
  v8h hi = *(const v8h*)(p + 8);
  return __builtin_shufflevector(lo, hi, 0,1,2,3,4,5,6,7,8,9,10,11,12,13,14,15);
}

// ---------------------------------------------------------------------------
// Kernel 1: convert W_enc/W_pred to f16 row-major; repack W_out (padded to
// 1152 rows) into WMMA-B fragment order:
//   wpack[(((n*20 + kc)*32 + lane)*16 + e]  where
//     v = n*16 + (lane&15), k = kc*32 + (lane>=16?8:0) + (e<8 ? e : 8+e)
// Each (n,kc) fragment block is 1KB contiguous -> fully coalesced B loads.
// ---------------------------------------------------------------------------
__global__ __launch_bounds__(256) void convert_weights(
    const float* __restrict__ Wenc, const float* __restrict__ Wpred,
    const float* __restrict__ Wout,
    _Float16* __restrict__ wenc16, _Float16* __restrict__ wpred16,
    _Float16* __restrict__ wpack) {
  const int N1 = 640 * 512;
  const int N2 = 640 * 640;
  const int N3 = 1152 * 640;  // 72 tiles * 20 kc * 32 lanes * 16 halves
  int i = blockIdx.x * 256 + threadIdx.x;
  if (i < N1) {
    wenc16[i] = (_Float16)Wenc[i];
  } else if (i < N1 + N2) {
    int j = i - N1;
    wpred16[j] = (_Float16)Wpred[j];
  } else if (i < N1 + N2 + N3) {
    int j = i - N1 - N2;
    int e    = j & 15;
    int lane = (j >> 4) & 31;
    int kc   = (j >> 9) % 20;
    int n    = j / (20 * 32 * 16);
    int v = n * 16 + (lane & 15);
    int k = kc * 32 + ((lane >> 4) & 1) * 8 + (e < 8 ? e : 8 + e);
    wpack[j] = (v < 1025) ? (_Float16)Wout[(size_t)v * 640 + k] : (_Float16)0.0f;
  }
}

// ---------------------------------------------------------------------------
// Kernel 2: projection GEMM  out16[m][n] = f16( sum_k A[m][k]*W[n][k] + bias[n] )
//   A[m][k] = src[b][k][inner]   (m = b*rowsPerB + inner)
//   32 rows / workgroup, N = 640 (40 tiles -> 5 per wave), K runtime (512/640).
// ---------------------------------------------------------------------------
__global__ __launch_bounds__(256) void proj_wmma(
    const float* __restrict__ src, const _Float16* __restrict__ W16,
    const float* __restrict__ bias, _Float16* __restrict__ out16,
    int K, int rowsPerB) {
  __shared__ __align__(16) _Float16 sA[32 * 648];

  const int tid = threadIdx.x;
  const int m0 = blockIdx.x * 32;
  const int b = m0 / rowsPerB;
  const int inner0 = m0 % rowsPerB;
  const int pitch = K + 8;
  const int kc = K >> 3;
  const float* srcb = src + (size_t)b * K * rowsPerB + inner0;

  for (int idx = tid; idx < 32 * kc; idx += 256) {
    int r = idx / kc;
    int k = (idx - r * kc) * 8;
    _Float16* d = &sA[r * pitch + k];
    #pragma unroll
    for (int e = 0; e < 8; ++e)
      d[e] = (_Float16)srcb[(size_t)(k + e) * rowsPerB + r];
  }
  __syncthreads();

  const int w = tid >> 5;
  const int lane = tid & 31;
  const int ln = lane & 15;
  const int l4 = (lane >> 4) & 1;

  v8f C[2][5];
  #pragma unroll
  for (int a = 0; a < 2; ++a)
    #pragma unroll
    for (int i = 0; i < 5; ++i)
      #pragma unroll
      for (int e = 0; e < 8; ++e) C[a][i][e] = 0.0f;

  for (int k0 = 0; k0 < K; k0 += 32) {
    int koff = k0 + l4 * 8;
    v16h A0 = make_frag(&sA[ln * pitch + koff]);
    v16h A1 = make_frag(&sA[(16 + ln) * pitch + koff]);
    #pragma unroll
    for (int i = 0; i < 5; ++i) {
      int v = (w * 5 + i) * 16 + ln;
      v16h Bt = make_frag(&W16[(size_t)v * K + koff]);
      C[0][i] = WMMA_F16(A0, Bt, C[0][i]);
      C[1][i] = WMMA_F16(A1, Bt, C[1][i]);
    }
  }

  #pragma unroll
  for (int a = 0; a < 2; ++a)
    #pragma unroll
    for (int i = 0; i < 5; ++i) {
      int v = (w * 5 + i) * 16 + ln;
      float bv = bias[v];
      #pragma unroll
      for (int j = 0; j < 8; ++j) {
        int m = m0 + a * 16 + j + 8 * l4;
        out16[(size_t)m * 640 + v] = (_Float16)(C[a][i][j] + bv);
      }
    }
}

// ---------------------------------------------------------------------------
// Kernel 3: fused joint GEMM + log-softmax.
//   rows = (b, t, u0..u0+31); h[r][k] = relu(f[b*T+t][k] + g[b*U+u][k]) in LDS.
//   N padded to 1152 (72 tiles = 9/wave); pad bias = -1e30 -> exp()=0.
//   B fragments read from wpack in fragment order (fully coalesced).
// ---------------------------------------------------------------------------
__global__ __launch_bounds__(256) void rnnt_joint_wmma(
    const _Float16* __restrict__ fws, const _Float16* __restrict__ gws,
    const _Float16* __restrict__ wpack, const float* __restrict__ b_out,
    float* __restrict__ out) {
  constexpr int PITCH = 648;  // 640 + 8 halves: 16B-aligned rows, no 64-bank stride
  __shared__ __align__(16) _Float16 hA[32 * PITCH];  // 41472 B
  __shared__ float red[8][32][16];                   // 16384 B
  __shared__ float rowmax[32];
  __shared__ float rowlse[32];

  const int tid = threadIdx.x;
  const int ublk = blockIdx.x & 3;
  const int t = (blockIdx.x >> 2) & 255;
  const int b = blockIdx.x >> 10;
  const int frow = b * 256 + t;
  const int grow0 = b * 128 + ublk * 32;

  // Build 32x640 f16 h-tile: relu(f + g)
  for (int idx = tid; idx < 32 * 80; idx += 256) {
    int r = idx / 80;
    int k = (idx - r * 80) * 8;
    v8h fv = *(const v8h*)(fws + (size_t)frow * 640 + k);
    v8h gv = *(const v8h*)(gws + (size_t)(grow0 + r) * 640 + k);
    v8h hv = fv + gv;
    #pragma unroll
    for (int e = 0; e < 8; ++e)
      hv[e] = hv[e] > (_Float16)0.0f ? hv[e] : (_Float16)0.0f;
    *(v8h*)(&hA[r * PITCH + k]) = hv;
  }
  __syncthreads();

  const int w = tid >> 5;
  const int lane = tid & 31;
  const int ln = lane & 15;
  const int l4 = (lane >> 4) & 1;

  v8f C[2][9];
  #pragma unroll
  for (int a = 0; a < 2; ++a)
    #pragma unroll
    for (int i = 0; i < 9; ++i)
      #pragma unroll
      for (int e = 0; e < 8; ++e) C[a][i][e] = 0.0f;

  // Per-lane base into packed B: wave w owns tiles n = w*9 .. w*9+8.
  const _Float16* wb = wpack + ((size_t)(w * 9) * 20 + 0) * 512 + (size_t)lane * 16;

  for (int kc = 0; kc < 20; ++kc) {
    int koff = kc * 32 + l4 * 8;
    v16h A0 = make_frag(&hA[ln * PITCH + koff]);
    v16h A1 = make_frag(&hA[(16 + ln) * PITCH + koff]);
    #pragma unroll
    for (int i = 0; i < 9; ++i) {
      v16h Bt = make_frag_packed(wb + (size_t)(i * 20 + kc) * 512);
      C[0][i] = WMMA_F16(A0, Bt, C[0][i]);
      C[1][i] = WMMA_F16(A1, Bt, C[1][i]);
    }
  }

  float biasv[9];
  #pragma unroll
  for (int i = 0; i < 9; ++i) {
    int v = (w * 9 + i) * 16 + ln;
    biasv[i] = (v < 1025) ? b_out[v] : -1.0e30f;
  }

  // --- row max ---
  #pragma unroll
  for (int a = 0; a < 2; ++a)
    #pragma unroll
    for (int j = 0; j < 8; ++j) {
      float m = -3.4e38f;
      #pragma unroll
      for (int i = 0; i < 9; ++i) {
        float x = C[a][i][j] + biasv[i];
        m = x > m ? x : m;
      }
      red[w][a * 16 + j + 8 * l4][ln] = m;
    }
  __syncthreads();
  if (tid < 32) {
    float m = -3.4e38f;
    for (int s = 0; s < 8; ++s)
      for (int c = 0; c < 16; ++c) {
        float x = red[s][tid][c];
        m = x > m ? x : m;
      }
    rowmax[tid] = m;
  }
  __syncthreads();

  // --- row sum(exp) ---
  #pragma unroll
  for (int a = 0; a < 2; ++a)
    #pragma unroll
    for (int j = 0; j < 8; ++j) {
      int row = a * 16 + j + 8 * l4;
      float rm = rowmax[row];
      float s = 0.0f;
      #pragma unroll
      for (int i = 0; i < 9; ++i)
        s += __expf(C[a][i][j] + biasv[i] - rm);
      red[w][row][ln] = s;
    }
  __syncthreads();
  if (tid < 32) {
    float s = 0.0f;
    for (int q = 0; q < 8; ++q)
      for (int c = 0; c < 16; ++c) s += red[q][tid][c];
    rowlse[tid] = __logf(s);
  }
  __syncthreads();

  // --- write log-softmax ---
  #pragma unroll
  for (int a = 0; a < 2; ++a)
    #pragma unroll
    for (int j = 0; j < 8; ++j) {
      int row = a * 16 + j + 8 * l4;
      float sub = rowmax[row] + rowlse[row];
      size_t rb = ((size_t)(b * 256 + t) * 128 + (ublk * 32 + row)) * 1025;
      #pragma unroll
      for (int i = 0; i < 9; ++i) {
        int v = (w * 9 + i) * 16 + ln;
        if (v < 1025) out[rb + v] = C[a][i][j] + biasv[i] - sub;
      }
    }
}

// ---------------------------------------------------------------------------
extern "C" void kernel_launch(void* const* d_in, const int* in_sizes, int n_in,
                              void* d_out, int out_size, void* d_ws, size_t ws_size,
                              hipStream_t stream) {
  const float* enc   = (const float*)d_in[0];  // [4,512,256]
  const float* dec   = (const float*)d_in[1];  // [4,640,128]
  const float* Wenc  = (const float*)d_in[2];  // [640,512]
  const float* benc  = (const float*)d_in[3];  // [640]
  const float* Wpred = (const float*)d_in[4];  // [640,640]
  const float* bpred = (const float*)d_in[5];  // [640]
  const float* Wout  = (const float*)d_in[6];  // [1025,640]
  const float* bout  = (const float*)d_in[7];  // [1025]
  float* out = (float*)d_out;                  // [4,256,128,1025]

  _Float16* wenc16  = (_Float16*)d_ws;            // 640*512
  _Float16* wpred16 = wenc16 + 640 * 512;         // 640*640
  _Float16* wpack   = wpred16 + 640 * 640;        // 1152*640 (fragment-packed)
  _Float16* fws     = wpack + 1152 * 640;         // 1024*640
  _Float16* gws     = fws + 1024 * 640;           // 512*640
  // total ws: ~4.7 MB

  const int totalc = 640 * 512 + 640 * 640 + 1152 * 640;
  convert_weights<<<(totalc + 255) / 256, 256, 0, stream>>>(
      Wenc, Wpred, Wout, wenc16, wpred16, wpack);

  // f = enc^T @ Wenc^T + b_enc : M=1024, K=512, rowsPerB=T=256
  proj_wmma<<<32, 256, 0, stream>>>(enc, wenc16, benc, fws, 512, 256);
  // g = dec^T @ Wpred^T + b_pred : M=512, K=640, rowsPerB=U=128
  proj_wmma<<<16, 256, 0, stream>>>(dec, wpred16, bpred, gws, 640, 128);

  // fused joint GEMM + log-softmax: 4096 workgroups (B*T*U/32)
  rnnt_joint_wmma<<<4096, 256, 0, stream>>>(fws, gws, wpack, bout, out);
}
